// FractalDecoderV7_19645180412075
// MI455X (gfx1250) — matrix operand order repaired and checked by reading, not compiled, not run on hardware
//
#include <hip/hip_runtime.h>
#include <hip/hip_bf16.h>
#include <stdint.h>

typedef __attribute__((ext_vector_type(16))) _Float16 v16h;
typedef __attribute__((ext_vector_type(8)))  _Float16 v8h;
typedef __attribute__((ext_vector_type(8)))  float    v8f;

#define HS 264   // h LDS row stride in halfs (256 + 8 pad -> bank offset 4/row)
#define GS 520   // g LDS row stride (512 + 8)
#define FS 136   // feats LDS row stride (128 + 8)

static __device__ __forceinline__ v16h cat8(v8h lo, v8h hi) {
  return __builtin_shufflevector(lo, hi, 0,1,2,3,4,5,6,7,8,9,10,11,12,13,14,15);
}

static __device__ __forceinline__ float geluf(float v) {
  return 0.5f * v * (1.0f + erff(v * 0.70710678118654752440f));
}

// Wave-level GEMM: MT m-tiles (rows 16*MT), NPW n-tiles for this wave, KC k-chunks of 32.
// A: LDS f16, row-major with stride aStride (halfs). B: global f16 in fragment layout
// ((nt*ktTot + kt)*32 + lane)*16 + e, where e<->K linear per 16-lane half (lanes>=16: K+16).
template<int MT, int NPW, int KC>
static __device__ __forceinline__ void wave_gemm(v8f (&acc)[MT][NPW],
    const _Float16* A, int aStride,
    const _Float16* __restrict__ Bfrag, int ntBase, int ktTot, int lane)
{
  const int lhalf = (lane < 16) ? 0 : 16;
  const int arow  = lane & 15;
#pragma unroll
  for (int k = 0; k < KC; ++k) {
    v16h b[NPW];
#pragma unroll
    for (int j = 0; j < NPW; ++j)
      b[j] = *(const v16h*)(Bfrag + (((size_t)(ntBase + j) * ktTot + k) * 32 + lane) * 16);
    v16h a[MT];
#pragma unroll
    for (int i = 0; i < MT; ++i) {
      // A-frag (16x32 f16): lane<16 row m holds K {0..15}; lanes>=16 hold K {16..31}
      const _Float16* p = A + (size_t)(16 * i + arow) * aStride + 32 * k + lhalf;
      a[i] = cat8(*(const v8h*)p, *(const v8h*)(p + 8));
    }
#pragma unroll
    for (int i = 0; i < MT; ++i)
#pragma unroll
      for (int j = 0; j < NPW; ++j)
        acc[i][j] = __builtin_amdgcn_wmma_f32_16x16x32_f16(
            false, a[i], false, b[j], (short)0, acc[i][j], false, false);
  }
}

// ---------------- weight -> f16 B-fragment layout ----------------
__global__ void k_prep(const float* __restrict__ W, _Float16* __restrict__ out,
                       int K, int N, int Kpad) {
  size_t total = (size_t)(N >> 4) * (size_t)(Kpad >> 5) * 512;
  size_t idx = (size_t)blockIdx.x * blockDim.x + threadIdx.x;
  if (idx >= total) return;
  int e    = (int)(idx & 15);
  int lane = (int)((idx >> 4) & 31);
  size_t t = idx >> 9;
  int KT = Kpad >> 5;
  int kt = (int)(t % KT);
  int nt = (int)(t / KT);
  int k = kt * 32 + e + ((lane < 16) ? 0 : 16);
  int n = nt * 16 + (lane & 15);
  float v = (k < K) ? W[(size_t)k * N + n] : 0.0f;
  out[idx] = (_Float16)v;
}

// ---------------- feature build: fourier + hashgrid + latent -> f16 [8192][128] ----------------
__global__ void k_features(const float* __restrict__ x, const float* __restrict__ z,
                           const float* __restrict__ emb, _Float16* __restrict__ feats) {
  int b = blockIdx.x * blockDim.x + threadIdx.x;
  if (b >= 8192) return;
  float xc = x[b];
  xc = fminf(fmaxf(xc, 0.0f), 1.0f);
  _Float16* row = feats + (size_t)b * 128;
  row[0] = (_Float16)xc;
  double f = 1.0;
#pragma unroll 1
  for (int k = 0; k < 32; ++k) {
    float fr = (float)f;                 // float32-exact 3.5^k
    float arg = 6.28318530717958647692f * xc * fr;
    row[1 + k]  = (_Float16)sinf(arg);
    row[33 + k] = (_Float16)cosf(arg);
    f *= 3.5;
  }
#pragma unroll 1
  for (int l = 0; l < 12; ++l) {
    float res = (float)(16 << l);
    float pos = xc * res;
    float i0f = floorf(pos);
    int   i0  = (int)i0f;
    float w   = pos - i0f;
    int h0 = i0 & 16383;
    int h1 = (i0 + 1) & 16383;
    float e00 = emb[(size_t)h0 * 24 + 2 * l];
    float e01 = emb[(size_t)h0 * 24 + 2 * l + 1];
    float e10 = emb[(size_t)h1 * 24 + 2 * l];
    float e11 = emb[(size_t)h1 * 24 + 2 * l + 1];
    row[65 + 2 * l]     = (_Float16)((1.0f - w) * e00 + w * e10);
    row[65 + 2 * l + 1] = (_Float16)((1.0f - w) * e01 + w * e11);
  }
  for (int j = 0; j < 32; ++j) row[89 + j] = (_Float16)z[(size_t)b * 32 + j];
  for (int j = 121; j < 128; ++j) row[j] = (_Float16)0.0f;
}

// ---------------- main: mlp_in + 48 fractal steps, all in LDS ----------------
__global__ __launch_bounds__(256) void k_main(
    const _Float16* __restrict__ feats,
    const _Float16* __restrict__ fW1, const float* __restrict__ b1,
    const _Float16* __restrict__ fW2, const float* __restrict__ b2,
    const _Float16* __restrict__ fWf1, const float* __restrict__ bf1,
    const _Float16* __restrict__ fWf2, const float* __restrict__ bf2,
    _Float16* __restrict__ hOut)
{
  __shared__ _Float16 sH[64 * HS];   // 33792 B
  __shared__ _Float16 sG[64 * GS];   // 66560 B
  const int tid  = threadIdx.x;
  const int lane = tid & 31, wave = tid >> 5;
  const int hi8  = (lane < 16) ? 0 : 8;
  const int n15  = lane & 15;
  const int row0 = blockIdx.x * 64;

  // stage feats tile [64][128] into sH (stride FS)
  {
    const uint32_t* src = (const uint32_t*)(feats + (size_t)row0 * 128);
    uint32_t* dst = (uint32_t*)sH;
    for (int i = tid; i < 64 * 64; i += 256) {
      int r = i >> 6, c = i & 63;
      dst[r * (FS / 2) + c] = src[r * 64 + c];
    }
  }
  __syncthreads();

  // GEMM0: feats @ W_in1 -> gelu -> sG (h-layout, stride HS)
  {
    v8f acc[4][2];
    for (int i = 0; i < 4; ++i) for (int j = 0; j < 2; ++j)
      for (int r = 0; r < 8; ++r) acc[i][j][r] = 0.0f;
    wave_gemm<4, 2, 4>(acc, sH, FS, fW1, wave * 2, 4, lane);
    for (int i = 0; i < 4; ++i)
      for (int j = 0; j < 2; ++j) {
        int col = wave * 32 + j * 16 + n15;
        float bias = b1[col];
        for (int r = 0; r < 8; ++r) {
          int m = 16 * i + r + hi8;
          sG[m * HS + col] = (_Float16)geluf(acc[i][j][r] + bias);
        }
      }
  }
  __syncthreads();

  // GEMM0b: sG @ W_in2 -> gelu -> sH (canonical h)
  {
    v8f acc[4][2];
    for (int i = 0; i < 4; ++i) for (int j = 0; j < 2; ++j)
      for (int r = 0; r < 8; ++r) acc[i][j][r] = 0.0f;
    wave_gemm<4, 2, 8>(acc, sG, HS, fW2, wave * 2, 8, lane);
    for (int i = 0; i < 4; ++i)
      for (int j = 0; j < 2; ++j) {
        int col = wave * 32 + j * 16 + n15;
        float bias = b2[col];
        for (int r = 0; r < 8; ++r) {
          int m = 16 * i + r + hi8;
          sH[m * HS + col] = (_Float16)geluf(acc[i][j][r] + bias);
        }
      }
  }
  __syncthreads();

  // hoist per-lane biases for the fractal loop
  float rbf1[4], rbf2[2];
  for (int j = 0; j < 4; ++j) rbf1[j] = bf1[wave * 64 + j * 16 + n15];
  for (int j = 0; j < 2; ++j) rbf2[j] = bf2[wave * 32 + j * 16 + n15];

#pragma unroll 1
  for (int s = 0; s < 48; ++s) {
    // g = gelu(h @ Wf1 + bf1) : [64,256] x [256,512]
    {
      v8f acc[4][4];
      for (int i = 0; i < 4; ++i) for (int j = 0; j < 4; ++j)
        for (int r = 0; r < 8; ++r) acc[i][j][r] = 0.0f;
      wave_gemm<4, 4, 8>(acc, sH, HS, fWf1, wave * 4, 8, lane);
      for (int i = 0; i < 4; ++i)
        for (int j = 0; j < 4; ++j) {
          int col = wave * 64 + j * 16 + n15;
          for (int r = 0; r < 8; ++r) {
            int m = 16 * i + r + hi8;
            sG[m * GS + col] = (_Float16)geluf(acc[i][j][r] + rbf1[j]);
          }
        }
    }
    __syncthreads();
    // h = 0.5*h + 0.5*tanh(g @ Wf2 + bf2) : [64,512] x [512,256]
    {
      v8f acc[4][2];
      for (int i = 0; i < 4; ++i) for (int j = 0; j < 2; ++j)
        for (int r = 0; r < 8; ++r) acc[i][j][r] = 0.0f;
      wave_gemm<4, 2, 16>(acc, sG, GS, fWf2, wave * 2, 16, lane);
      for (int i = 0; i < 4; ++i)
        for (int j = 0; j < 2; ++j) {
          int col = wave * 32 + j * 16 + n15;
          for (int r = 0; r < 8; ++r) {
            int m = 16 * i + r + hi8;
            float u  = tanhf(acc[i][j][r] + rbf2[j]);
            float ho = (float)sH[m * HS + col];
            sH[m * HS + col] = (_Float16)(0.5f * (ho + u));
          }
        }
    }
    __syncthreads();
  }

  // write h tile back to global f16 [8192][256]
  {
    uint32_t* dst = (uint32_t*)(hOut + (size_t)row0 * 256);
    const uint32_t* src = (const uint32_t*)sH;
    for (int i = tid; i < 64 * 128; i += 256) {
      int r = i >> 7, c = i & 127;
      dst[r * 128 + c] = src[r * (HS / 2) + c];
    }
  }
}

// ---------------- output head: logits = h @ W_out + b_out (fp32 out) ----------------
__global__ __launch_bounds__(256) void k_out(
    const _Float16* __restrict__ hG,
    const _Float16* __restrict__ fWout, const float* __restrict__ bout,
    float* __restrict__ out)
{
  __shared__ _Float16 sH[64 * HS];
  const int tid = threadIdx.x, lane = tid & 31, wave = tid >> 5;
  const int hi8 = (lane < 16) ? 0 : 8, n15 = lane & 15;
  const int row0 = blockIdx.y * 64;
  const int nb = blockIdx.x;          // 125 blocks of 256 cols
  {
    const uint32_t* src = (const uint32_t*)(hG + (size_t)row0 * 256);
    uint32_t* dst = (uint32_t*)sH;
    for (int i = tid; i < 64 * 128; i += 256) {
      int r = i >> 7, c = i & 127;
      dst[r * (HS / 2) + c] = src[r * 128 + c];
    }
  }
  __syncthreads();
  v8f acc[4][2];
  for (int i = 0; i < 4; ++i) for (int j = 0; j < 2; ++j)
    for (int r = 0; r < 8; ++r) acc[i][j][r] = 0.0f;
  wave_gemm<4, 2, 8>(acc, sH, HS, fWout, nb * 16 + wave * 2, 8, lane);
  for (int i = 0; i < 4; ++i)
    for (int j = 0; j < 2; ++j) {
      int col = nb * 256 + wave * 32 + j * 16 + n15;
      float bias = bout[col];
      for (int r = 0; r < 8; ++r) {
        int m = row0 + 16 * i + r + hi8;
        out[(size_t)m * 32000 + col] = acc[i][j][r] + bias;
      }
    }
}

extern "C" void kernel_launch(void* const* d_in, const int* in_sizes, int n_in,
                              void* d_out, int out_size, void* d_ws, size_t ws_size,
                              hipStream_t stream) {
  const float* x    = (const float*)d_in[0];
  const float* z    = (const float*)d_in[1];
  const float* emb  = (const float*)d_in[2];
  const float* Win1 = (const float*)d_in[3];
  const float* bin1 = (const float*)d_in[4];
  const float* Win2 = (const float*)d_in[5];
  const float* bin2 = (const float*)d_in[6];
  const float* Wf1  = (const float*)d_in[7];
  const float* bf1  = (const float*)d_in[8];
  const float* Wf2  = (const float*)d_in[9];
  const float* bf2  = (const float*)d_in[10];
  const float* Wout = (const float*)d_in[11];
  const float* bout = (const float*)d_in[12];
  float* out = (float*)d_out;

  char* ws = (char*)d_ws;
  _Float16* feats = (_Float16*)(ws);               // 8192*128*2   = 2 MB
  _Float16* hbuf  = (_Float16*)(ws + 2097152);     // 8192*256*2   = 4 MB
  _Float16* fW1   = (_Float16*)(ws + 6291456);     // 64 KB
  _Float16* fW2   = (_Float16*)(ws + 6356992);     // 128 KB
  _Float16* fWf1  = (_Float16*)(ws + 6488064);     // 256 KB
  _Float16* fWf2  = (_Float16*)(ws + 6750208);     // 256 KB
  _Float16* fWo   = (_Float16*)(ws + 7012352);     // 16 MB

  auto nb = [](size_t total) { return (unsigned)((total + 255) / 256); };
  k_prep<<<nb((size_t)16 * 4 * 512),    256, 0, stream>>>(Win1, fW1, 121, 256, 128);
  k_prep<<<nb((size_t)16 * 8 * 512),    256, 0, stream>>>(Win2, fW2, 256, 256, 256);
  k_prep<<<nb((size_t)32 * 8 * 512),    256, 0, stream>>>(Wf1,  fWf1, 256, 512, 256);
  k_prep<<<nb((size_t)16 * 16 * 512),   256, 0, stream>>>(Wf2,  fWf2, 512, 256, 512);
  k_prep<<<nb((size_t)2000 * 8 * 512),  256, 0, stream>>>(Wout, fWo,  256, 32000, 256);
  k_features<<<32, 256, 0, stream>>>(x, z, emb, feats);
  k_main<<<128, 256, 0, stream>>>(feats, fW1, bin1, fW2, bin2, fWf1, bf1, fWf2, bf2, hbuf);
  k_out<<<dim3(125, 128), 256, 0, stream>>>(hbuf, fWo, bout, out);
}
